// NonLocalBlock_28630251995431
// MI455X (gfx1250) — compile-verified, matrix operand order checked by reading, not run on hardware
//
#include <hip/hip_runtime.h>

// ---------------------------------------------------------------------------
// NonLocalBlock for MI455X (gfx1250, wave32, WMMA bf16 -> f32 accum).
// B=4, C=512, IC=128, N=16*14*14=3136 = 49*64.
// Software-pipelined fragment loads: next fragment's global loads are issued
// before the current WMMA so waits become partial instead of full drains.
// ---------------------------------------------------------------------------

typedef __bf16 bf16_t;
typedef __attribute__((ext_vector_type(16))) __bf16 v16bf;
typedef __attribute__((ext_vector_type(8)))  __bf16 v8bf;
typedef __attribute__((ext_vector_type(8)))  float  v8f;

__device__ __forceinline__ v8f wmma_bf16(v16bf a, v16bf b, v8f c) {
    // D = A(16x32) * B(32x16) + C(16x16 f32)
    return __builtin_amdgcn_wmma_f32_16x16x32_bf16(false, a, false, b, (short)0, c,
                                                   false, false);
}

// Load one 16x32 bf16 fragment (A layout; B uses identical striping on B^T rows).
// Per CDNA5 ISA 7.12.2: lane m=lane&15, half=lane>>4; VGPR0-3 hold K=8h..8h+7,
// VGPR4-7 hold K=16+8h..16+8h+7  ->  two contiguous 16-byte loads per lane.
__device__ __forceinline__ v16bf load_frag(const bf16_t* base, int stride, int lane) {
    const int r = lane & 15;
    const int h = lane >> 4;
    const bf16_t* p = base + (long)r * stride + 8 * h;
    v8bf lo = *reinterpret_cast<const v8bf*>(p);
    v8bf hi = *reinterpret_cast<const v8bf*>(p + 16);
    v16bf v;
#pragma unroll
    for (int i = 0; i < 8; ++i) { v[i] = lo[i]; v[i + 8] = hi[i]; }
    return v;
}

// ------------------------- constants ---------------------------------------
#define BB 4
#define CC 512
#define ICC 128
#define NN 3136            // 49 * 64
#define NPART 16           // partial-reduction blocks for BN stats

// ------------------- kernel 0a: maps (b,c,n) -> mapsT bf16 (b,n,c) ---------
__global__ void k_cast_maps(const float* __restrict__ maps, bf16_t* __restrict__ mapsT) {
    __shared__ float tile[32][33];
    const int b  = blockIdx.z;
    const int c0 = blockIdx.y * 32;
    const int n0 = blockIdx.x * 32;
    const int tx = threadIdx.x, ty = threadIdx.y;   // block (32,8)
#pragma unroll
    for (int i = 0; i < 32; i += 8)
        tile[ty + i][tx] = maps[((long)b * CC + c0 + ty + i) * NN + n0 + tx];
    __syncthreads();
#pragma unroll
    for (int i = 0; i < 32; i += 8)
        mapsT[((long)b * NN + n0 + ty + i) * CC + c0 + tx] = (bf16_t)tile[tx][ty + i];
}

// ------------------- kernel 0b: generic f32 -> bf16 cast -------------------
__global__ void k_cast(const float* __restrict__ in, bf16_t* __restrict__ out, int n) {
    int i = blockIdx.x * blockDim.x + threadIdx.x;
    if (i < n) out[i] = (bf16_t)in[i];
}

// ------------------- kernel 1: single-output 1x1 conv GEMM -----------------
// Out^T[b][n][ic] = mapsT[b][n][c] * w[ic][c] + bias[ic].
// grid (49, B), block 128 (4 waves x 16 rows = 64 positions). 8 f32 accums/wave.
__global__ __launch_bounds__(128, 1)
void k_conv1(const bf16_t* __restrict__ mapsT, const bf16_t* __restrict__ w,
             const float* __restrict__ bias, float* __restrict__ outT) {
    const int b    = blockIdx.y;
    const int lane = threadIdx.x & 31;
    const int wave = threadIdx.x >> 5;
    const int row0 = blockIdx.x * 64 + wave * 16;
    const bf16_t* Abase = mapsT + ((long)b * NN + row0) * CC;

    v8f acc[8];
#pragma unroll
    for (int j = 0; j < 8; ++j)
#pragma unroll
        for (int e = 0; e < 8; ++e) acc[j][e] = 0.f;

    // pipelined: fragment loads for step i+1 issued before WMMA of step i
    v16bf a_cur = load_frag(Abase, CC, lane);
    v16bf b_cur = load_frag(w, CC, lane);
    for (int k0 = 0; k0 < CC; k0 += 32) {
        v16bf a_next;
        if (k0 + 32 < CC) a_next = load_frag(Abase + k0 + 32, CC, lane);
#pragma unroll
        for (int j = 0; j < 8; ++j) {
            v16bf b_nx;
            if (j < 7)
                b_nx = load_frag(w + (long)((j + 1) * 16) * CC + k0, CC, lane);
            else if (k0 + 32 < CC)
                b_nx = load_frag(w + k0 + 32, CC, lane);
            acc[j] = wmma_bf16(a_cur, b_cur, acc[j]);
            b_cur = b_nx;
        }
        a_cur = a_next;
    }
    const int col = lane & 15, h = lane >> 4;
#pragma unroll
    for (int j = 0; j < 8; ++j) {
        const int ic = j * 16 + col;
        const float bv = bias[ic];
#pragma unroll
        for (int r = 0; r < 8; ++r)
            outT[((long)b * NN + row0 + r + 8 * h) * ICC + ic] = acc[j][r] + bv;
    }
}

// ------------------- kernel 2a: BN stats, per-block partials ---------------
// grid (NPART, 2), block 256.  Lane = channel -> fully coalesced row reads.
// part layout: [blk][which][ic] -> (sum, sumsq).  Deterministic (no atomics).
__global__ void k_stats_partial(const float* __restrict__ thetaT,
                                const float* __restrict__ globT,
                                float* __restrict__ part) {
    const int which = blockIdx.y;
    const float* src = which ? globT : thetaT;
    const int ic   = threadIdx.x & (ICC - 1);
    const int half = threadIdx.x >> 7;            // two rows per block-iteration
    const int R = BB * NN;                        // 12544 rows
    float s = 0.f, s2 = 0.f;
    for (int r = blockIdx.x * 2 + half; r < R; r += NPART * 2) {
        float v = src[(long)r * ICC + ic];
        s += v; s2 += v * v;
    }
    __shared__ float sh[256];
    sh[threadIdx.x] = s;
    __syncthreads();
    if (threadIdx.x < ICC)
        part[((blockIdx.x * 2 + which) * ICC + threadIdx.x) * 2 + 0] =
            sh[threadIdx.x] + sh[threadIdx.x + ICC];
    __syncthreads();
    sh[threadIdx.x] = s2;
    __syncthreads();
    if (threadIdx.x < ICC)
        part[((blockIdx.x * 2 + which) * ICC + threadIdx.x) * 2 + 1] =
            sh[threadIdx.x] + sh[threadIdx.x + ICC];
}

// ------------------- kernel 2b: finalize mean/rstd -------------------------
__global__ void k_stats_final(const float* __restrict__ part, float* __restrict__ stats) {
    const int i = threadIdx.x;                    // 0..255
    if (i >= 2 * ICC) return;
    const int which = i >> 7, ic = i & (ICC - 1);
    float s = 0.f, s2 = 0.f;
    for (int blk = 0; blk < NPART; ++blk) {
        s  += part[((blk * 2 + which) * ICC + ic) * 2 + 0];
        s2 += part[((blk * 2 + which) * ICC + ic) * 2 + 1];
    }
    const float inv = 1.0f / (float)(BB * NN);
    float mean = s * inv;
    float var  = s2 * inv - mean * mean;
    stats[i * 2 + 0] = mean;
    stats[i * 2 + 1] = rsqrtf(var + 1e-5f);
}

// ------------------- kernel 3: normalize + cast (tiled) --------------------
// theta_bf: pos-major [b][n][ic] (straight-through, coalesced).
// glob_dm:  dim-major [b][ic][n] via 32x32 LDS transpose (coalesced writes).
__global__ void k_norm(const float* __restrict__ thetaT, const float* __restrict__ globT,
                       const float* __restrict__ stats,
                       bf16_t* __restrict__ theta_bf, bf16_t* __restrict__ glob_dm) {
    __shared__ float gt[32][33];
    const int b   = blockIdx.z;
    const int ic0 = blockIdx.y * 32;
    const int n0  = blockIdx.x * 32;
    const int tx = threadIdx.x, ty = threadIdx.y;   // block (32,8)
    const int icx = ic0 + tx;
    const float mt = stats[icx * 2 + 0], rt = stats[icx * 2 + 1];
#pragma unroll
    for (int i = 0; i < 32; i += 8) {
        const long idx = ((long)b * NN + n0 + ty + i) * ICC + icx;
        theta_bf[idx] = (bf16_t)((thetaT[idx] - mt) * rt);
        gt[ty + i][tx] = globT[idx];
    }
    __syncthreads();
#pragma unroll
    for (int i = 0; i < 32; i += 8) {
        const int ic = ic0 + ty + i;
        const float mg = stats[(ICC + ic) * 2 + 0], rg = stats[(ICC + ic) * 2 + 1];
        glob_dm[((long)b * ICC + ic) * NN + n0 + tx] = (bf16_t)((gt[tx][ty + i] - mg) * rg);
    }
}

// ------------------- kernel 4: streaming "attention" -----------------------
// relation = theta^T theta / n, softmax rows, y = softmax * glob^T.
// No running max needed: |S|/n << 1 so exp never overflows; normalize by row
// sum at the end.  grid (49, B), block 128 (4 waves, 64 query rows).
__global__ __launch_bounds__(128, 1)
void k_attn(const bf16_t* __restrict__ theta_bf,
            const bf16_t* __restrict__ glob_dm,
            bf16_t* __restrict__ y_bf) {
    constexpr int TK = 64;
    const float inv_n = 1.0f / (float)NN;
    const int b    = blockIdx.y;
    const int lane = threadIdx.x & 31;
    const int wave = threadIdx.x >> 5;
    const int qrow = blockIdx.x * 64 + wave * 16;

    __shared__ __align__(16) bf16_t Pl[64 * TK];     // wave-private 16xTK strips
    bf16_t* Pw = Pl + wave * 16 * TK;

    const bf16_t* tb = theta_bf + (long)b * NN * ICC;   // [n][ic]
    const bf16_t* vb = glob_dm  + (long)b * ICC * NN;   // [ic][n]

    // Q fragments: 4 K-chunks of 32, resident for the whole kernel.
    v16bf qf[4];
#pragma unroll
    for (int c = 0; c < 4; ++c)
        qf[c] = load_frag(tb + (long)qrow * ICC + c * 32, ICC, lane);

    v8f O[8];
    float lsum[8];
#pragma unroll
    for (int t = 0; t < 8; ++t) {
#pragma unroll
        for (int e = 0; e < 8; ++e) O[t][e] = 0.f;
        lsum[t] = 0.f;
    }

    const int col = lane & 15, h = lane >> 4;
    for (int kt = 0; kt < NN; kt += TK) {
        // hint-prefetch next key tile's theta and glob rows (global_prefetch_b8)
        if (kt + TK < NN) {
            __builtin_prefetch(tb + (long)(kt + TK) * ICC + (threadIdx.x & 127) * 32, 0, 0);
            __builtin_prefetch(vb + kt + TK + ((threadIdx.x & 127) >> 2) * NN +
                               (threadIdx.x & 3) * 16, 0, 0);
        }
        // S(16 x TK) = Q * K^T, pipelined K-fragment loads
        const bf16_t* kb = tb + (long)kt * ICC;
        v8f S[4];
#pragma unroll
        for (int j = 0; j < 4; ++j)
#pragma unroll
            for (int e = 0; e < 8; ++e) S[j][e] = 0.f;

        v16bf kf_cur = load_frag(kb, ICC, lane);
#pragma unroll
        for (int j = 0; j < 4; ++j) {
#pragma unroll
            for (int c = 0; c < 4; ++c) {
                v16bf kf_nx;
                const int idx = j * 4 + c;
                if (idx < 15) {
                    const int jn = (idx + 1) >> 2, cn = (idx + 1) & 3;
                    kf_nx = load_frag(kb + (long)(jn * 16) * ICC + cn * 32, ICC, lane);
                }
                S[j] = wmma_bf16(qf[c], kf_cur, S[j]);
                kf_cur = kf_nx;
            }
        }
        // P = exp(S/n); row-sum partials; C-layout -> A-layout via wave-private LDS
#pragma unroll
        for (int j = 0; j < 4; ++j)
#pragma unroll
            for (int r = 0; r < 8; ++r) {
                float p = __expf(S[j][r] * inv_n);
                lsum[r] += p;
                Pw[(r + 8 * h) * TK + j * 16 + col] = (bf16_t)p;
            }
        // O += P * V, pipelined V-fragment loads
#pragma unroll
        for (int c2 = 0; c2 < 2; ++c2) {
            v16bf pf = load_frag(Pw + c2 * 32, TK, lane);
            v16bf vf_cur = load_frag(vb + kt + c2 * 32, NN, lane);
#pragma unroll
            for (int t = 0; t < 8; ++t) {
                v16bf vf_nx;
                if (t < 7)
                    vf_nx = load_frag(vb + (long)((t + 1) * 16) * NN + kt + c2 * 32, NN, lane);
                O[t] = wmma_bf16(pf, vf_cur, O[t]);
                vf_cur = vf_nx;
            }
        }
    }

    // reduce row sums across the 16 lanes of each half (masks 1,2,4,8 stay in-half)
#pragma unroll
    for (int r = 0; r < 8; ++r) {
        float v = lsum[r];
#pragma unroll
        for (int m = 8; m >= 1; m >>= 1) v += __shfl_xor(v, m, 32);
        lsum[r] = v;
    }
    // y_bf pos-major [b][q][d]
#pragma unroll
    for (int t = 0; t < 8; ++t) {
        const int d = t * 16 + col;
#pragma unroll
        for (int r = 0; r < 8; ++r) {
            const int q = qrow + r + 8 * h;
            y_bf[((long)b * NN + q) * ICC + d] = (bf16_t)(O[t][r] / lsum[r]);
        }
    }
}

// ------------------- kernel 5: up-conv + bias + residual -------------------
// Out[b][co][n] = w_up[co][ic]*y[n][ic] + b_up[co] + maps[b][co][n].
// WMMA result staged through LDS so the [co][pos] stores (and the residual
// maps reads) are contiguous 64B runs.  grid (49, 4, B), block 128.
__global__ __launch_bounds__(128, 1)
void k_up(const bf16_t* __restrict__ y_bf, const bf16_t* __restrict__ wup,
          const float* __restrict__ bup, const float* __restrict__ maps,
          float* __restrict__ out) {
    __shared__ float ot[128][65];                 // [co within tile][pos within tile]
    const int b    = blockIdx.z;
    const int co0  = blockIdx.y * 128;
    const int p0   = blockIdx.x * 64;
    const int lane = threadIdx.x & 31;
    const int wave = threadIdx.x >> 5;
    const int row0 = p0 + wave * 16;
    const bf16_t* Ab = y_bf + ((long)b * NN + row0) * ICC;
    const bf16_t* Wb = wup + (long)co0 * ICC;

    v8f acc[8];
#pragma unroll
    for (int j = 0; j < 8; ++j)
#pragma unroll
        for (int e = 0; e < 8; ++e) acc[j][e] = 0.f;

    v16bf a_cur = load_frag(Ab, ICC, lane);
    v16bf b_cur = load_frag(Wb, ICC, lane);
#pragma unroll
    for (int c = 0; c < 4; ++c) {
        v16bf a_next;
        if (c < 3) a_next = load_frag(Ab + (c + 1) * 32, ICC, lane);
#pragma unroll
        for (int j = 0; j < 8; ++j) {
            v16bf b_nx;
            if (j < 7)
                b_nx = load_frag(Wb + (long)((j + 1) * 16) * ICC + c * 32, ICC, lane);
            else if (c < 3)
                b_nx = load_frag(Wb + (c + 1) * 32, ICC, lane);
            acc[j] = wmma_bf16(a_cur, b_cur, acc[j]);
            b_cur = b_nx;
        }
        a_cur = a_next;
    }
    const int col = lane & 15, h = lane >> 4;
#pragma unroll
    for (int j = 0; j < 8; ++j)
#pragma unroll
        for (int r = 0; r < 8; ++r)
            ot[j * 16 + col][wave * 16 + r + 8 * h] = acc[j][r];
    __syncthreads();

    // 512 chunks of (co, 16 consecutive positions); 4 per thread.
#pragma unroll
    for (int it = 0; it < 4; ++it) {
        const int idx = threadIdx.x + it * 128;   // 0..511
        const int co  = idx >> 2;                 // 0..127
        const int p   = (idx & 3) * 16;           // 0,16,32,48
        const long obase = ((long)b * CC + co0 + co) * NN + p0 + p;
        const float bias = bup[co0 + co];
        const float4* mp = reinterpret_cast<const float4*>(maps + obase);
        float4* op = reinterpret_cast<float4*>(out + obase);
#pragma unroll
        for (int q = 0; q < 4; ++q) {
            float4 m = mp[q];
            float4 r;
            r.x = ot[co][p + 4 * q + 0] + bias + m.x;
            r.y = ot[co][p + 4 * q + 1] + bias + m.y;
            r.z = ot[co][p + 4 * q + 2] + bias + m.z;
            r.w = ot[co][p + 4 * q + 3] + bias + m.w;
            op[q] = r;
        }
    }
}

// ---------------------------------------------------------------------------
extern "C" void kernel_launch(void* const* d_in, const int* in_sizes, int n_in,
                              void* d_out, int out_size, void* d_ws, size_t ws_size,
                              hipStream_t stream) {
    (void)in_sizes; (void)n_in; (void)out_size; (void)ws_size;
    const float* maps    = (const float*)d_in[0];
    const float* w_theta = (const float*)d_in[1];
    const float* b_theta = (const float*)d_in[2];
    // d_in[3], d_in[4] = w_phalt / b_phalt: dead code in the reference, skipped.
    const float* w_glob  = (const float*)d_in[5];
    const float* b_glob  = (const float*)d_in[6];
    const float* w_up    = (const float*)d_in[7];
    const float* b_up    = (const float*)d_in[8];
    float* out = (float*)d_out;

    char* ws = (char*)d_ws;
    size_t off = 0;
    auto carve = [&](size_t bytes) -> char* {
        char* p = ws + off;
        off = (off + bytes + 255) & ~(size_t)255;
        return p;
    };
    bf16_t* mapsT    = (bf16_t*)carve((size_t)BB * NN * CC * 2);   // 12.8 MB
    bf16_t* wth_bf   = (bf16_t*)carve((size_t)ICC * CC * 2);
    bf16_t* wgl_bf   = (bf16_t*)carve((size_t)ICC * CC * 2);
    bf16_t* wup_bf   = (bf16_t*)carve((size_t)CC * ICC * 2);
    float*  thetaT   = (float*) carve((size_t)BB * NN * ICC * 4);  // 6.4 MB
    float*  globT    = (float*) carve((size_t)BB * NN * ICC * 4);  // 6.4 MB
    float*  part     = (float*) carve((size_t)NPART * 2 * ICC * 2 * 4);
    float*  stats    = (float*) carve((size_t)2 * ICC * 2 * 4);
    bf16_t* theta_bf = (bf16_t*)carve((size_t)BB * NN * ICC * 2);  // 3.2 MB
    bf16_t* glob_dm  = (bf16_t*)carve((size_t)BB * NN * ICC * 2);  // 3.2 MB
    bf16_t* y_bf     = (bf16_t*)carve((size_t)BB * NN * ICC * 2);  // 3.2 MB

    // 0) casts / transposes
    k_cast_maps<<<dim3(NN / 32, CC / 32, BB), dim3(32, 8), 0, stream>>>(maps, mapsT);
    k_cast<<<dim3((ICC * CC + 255) / 256), dim3(256), 0, stream>>>(w_theta, wth_bf, ICC * CC);
    k_cast<<<dim3((ICC * CC + 255) / 256), dim3(256), 0, stream>>>(w_glob,  wgl_bf, ICC * CC);
    k_cast<<<dim3((CC * ICC + 255) / 256), dim3(256), 0, stream>>>(w_up,    wup_bf, CC * ICC);

    // 1) theta / glob 1x1 convs (WMMA, pipelined fragment loads)
    k_conv1<<<dim3(NN / 64, BB), dim3(128), 0, stream>>>(mapsT, wth_bf, b_theta, thetaT);
    k_conv1<<<dim3(NN / 64, BB), dim3(128), 0, stream>>>(mapsT, wgl_bf, b_glob,  globT);

    // 2) BN batch stats (deterministic two-stage reduction, coalesced)
    k_stats_partial<<<dim3(NPART, 2), dim3(256), 0, stream>>>(thetaT, globT, part);
    k_stats_final<<<dim3(1), dim3(256), 0, stream>>>(part, stats);

    // 3) normalize + cast (+ coalesced transpose of glob to dim-major)
    k_norm<<<dim3(NN / 32, ICC / 32, BB), dim3(32, 8), 0, stream>>>(thetaT, globT, stats,
                                                                    theta_bf, glob_dm);
    // 4) streaming relation/softmax/aggregate (WMMA, pipelined + prefetch)
    k_attn<<<dim3(NN / 64, BB), dim3(128), 0, stream>>>(theta_bf, glob_dm, y_bf);
    // 5) up-conv + bias + residual (WMMA, LDS-staged coalesced stores)
    k_up<<<dim3(NN / 64, 4, BB), dim3(128), 0, stream>>>(y_bf, wup_bf, b_up, maps, out);
}
